// ForwardSim_FeedbackOnlyTrialGLM_26448408609571
// MI455X (gfx1250) — matrix operand order repaired and checked by reading, not compiled, not run on hardware
//
#include <hip/hip_runtime.h>

// ---------------------------------------------------------------------------
// ForwardSim_FeedbackOnlyTrialGLM — MI455X (gfx1250) implementation
//
//   g[b,t]      = (X[b,:]·w_spat) * corr(stim_time, w_time)[t] + bias
//   spike[b,t]  = sigmoid(g[b,t] + dot(hist[b, t-250..t), w_fb))   (serial in t)
//   out1        = concat(init_hist, spikes) broadcast over 256 repeats
//   out2        = gensigs broadcast over 256 repeats
//
// All 256 repeats are identical (deterministic surrogate, repeat-independent
// init/g) -> simulate 128 unique sequences, then broadcast (360 MB of writes
// at 23.3 TB/s dominates, ~16 us).
//
// Per 16-step block the 250-tap feedback splits into
//   oldpart(16x16) = Hist(16x256, f16) @ Bshift(256x16, f16)   [8x WMMA]
// plus a <=15-FMA serial tail with register-resident tail weights.
// B fragments are loop-invariant registers; all 8 A fragments are preloaded
// per block so the WMMA chain overlaps the LDS loads (staggered s_wait_dscnt)
// instead of draining the DS pipe before every WMMA.
// ---------------------------------------------------------------------------

typedef __attribute__((ext_vector_type(16))) _Float16 v16h;
typedef __attribute__((ext_vector_type(8)))  _Float16 v8h;
typedef __attribute__((ext_vector_type(8)))  float    v8f;
typedef __attribute__((ext_vector_type(2)))  float    f2;

#define BATCH 128
#define NPIX  2048
#define NBINS 1500
#define NBF   250          // filter length
#define NREP  256
#define TSIM  1250         // NBINS - NBF
#define LENG  1251         // NBINS - NBF + 1 (valid correlation length)
#define BLK   16           // timesteps per WMMA block
#define NBLK  79           // ceil(TSIM / BLK)
#define KPAD  256          // K padded for WMMA (250 -> 256, zero-padded)
// Bank-conflict padding: row strides == 4 dwords (mod 64 banks) so the 16
// fragment lanes (one row/col each) start 4 banks apart = one b128 span.
#define HSTRIDE 1544       // halfs per H row (3088 B; 772 dwords; 772%64 == 4)
#define BSTRIDE 264        // halfs per BshT row (528 B; 132 dwords; 132%64 == 4)
#define OSTRIDE 20         // floats per oldp row (20*s % 64 distinct for s<16)

// ---------------------------------------------------------------------------
// Kernel A: filt_time[t] = sum_k stim_time[t+k]*w_time[k]   (t < 1251)
//           filt_spat[b] = sum_p X[b,p]*w_spat[p]           (b < 128)
// ---------------------------------------------------------------------------
__global__ __launch_bounds__(256) void precompute_kernel(
    const float* __restrict__ stim_time, const float* __restrict__ w_time,
    const float* __restrict__ X,         const float* __restrict__ w_spat,
    float* __restrict__ ft,              float* __restrict__ fs)
{
  int tid = blockIdx.x * 256 + threadIdx.x;
  if (tid < LENG) {
    float acc = 0.f;
    for (int k = 0; k < NBF; ++k) acc = fmaf(stim_time[tid + k], w_time[k], acc);
    ft[tid] = acc;
  } else if (tid >= 1280 && tid < 1280 + BATCH) {
    int b = tid - 1280;
    const float* row = X + (size_t)b * NPIX;
    float acc = 0.f;
    for (int p = 0; p < NPIX; ++p) acc = fmaf(row[p], w_spat[p], acc);
    fs[b] = acc;
  }
}

// ---------------------------------------------------------------------------
// Kernel B: simulate 16 batches per single-wave workgroup (8 WGs total).
// ---------------------------------------------------------------------------
__global__ __launch_bounds__(32) void sim_kernel(
    const float* __restrict__ init_sec,   // (BATCH, NBF)
    const float* __restrict__ fb_filter,  // (NBF)
    const float* __restrict__ ft,         // (LENG)   from ws
    const float* __restrict__ fs,         // (BATCH)  from ws
    const float* __restrict__ bias,       // (1)
    float* __restrict__ spike_ws,         // (BATCH, TSIM)
    float* __restrict__ gensig_ws)        // (BATCH, TSIM)
{
  __shared__ __align__(16) _Float16 H[16 * HSTRIDE];     // spike history, f16
  __shared__ __align__(16) _Float16 BshT[16 * BSTRIDE];  // BshT[n][i] = Bshift[i][n]
  __shared__ float oldp[16 * OSTRIDE];                   // WMMA result staging

  const int lane = threadIdx.x;
  const int b0   = blockIdx.x * 16;

  // --- zero history (padding taps multiply by zero weights; must not be NaN)
  {
    int4* hz = (int4*)&H[0];
    const int n4 = (16 * HSTRIDE * 2) / 16;
    for (int k = lane; k < n4; k += 32) hz[k] = make_int4(0, 0, 0, 0);
  }
  __syncthreads();

  // --- init history (identical for every repeat) as f16
  for (int e = lane; e < 16 * NBF; e += 32) {
    int r = e / NBF, i = e - r * NBF;
    H[r * HSTRIDE + i] = (_Float16)init_sec[(size_t)(b0 + r) * NBF + i];
  }
  // --- shifted-filter matrix, stored N-major for the B-fragment striping:
  //     Bshift[i][tau] = (i >= tau && i - tau < NBF && i < NBF) ? w[i-tau] : 0
  for (int e = lane; e < 16 * KPAD; e += 32) {
    int n = e >> 8, i = e & (KPAD - 1);
    int d = i - n;
    BshT[n * BSTRIDE + i] = (d >= 0 && d < NBF && i < NBF)
                                ? (_Float16)fb_filter[d] : (_Float16)0.f;
  }

  const float bias0 = bias[0];
  const float fs_b  = fs[b0 + (lane & 15)];

  // --- tail weights (w_fb[235..249]) live in registers for the serial part
  float wt[15];
#pragma unroll
  for (int k = 0; k < 15; ++k) wt[k] = fb_filter[NBF - 15 + k];
  __syncthreads();

  // gfx1250 16-bit WMMA fragment striping (wave32):
  //  A: lane L holds row M=L%16; K-octets {0..7,16..23} for L<16, {8..15,24..31}
  //  B: lane L holds col N=L%16; K-octets {0..7,16..23} for L<16, {8..15,24..31}
  const int arow  = lane & 15;
  const int akoff = (lane < 16) ? 0 : 8;
  const int bkoff = (lane < 16) ? 0 : 8;

  // --- B fragments are identical for every block: load once into registers
  v16h bfrag[8];
#pragma unroll
  for (int c = 0; c < 8; ++c) {
    v8h blo = *(const v8h*)&BshT[arow * BSTRIDE + 32 * c + bkoff];
    v8h bhi = *(const v8h*)&BshT[arow * BSTRIDE + 32 * c + bkoff + 16];
#pragma unroll
    for (int j = 0; j < 8; ++j) { bfrag[c][j] = blo[j]; bfrag[c][8 + j] = bhi[j]; }
  }

  for (int blk = 0; blk < NBLK; ++blk) {
    const int T0 = blk * BLK;

    // ---- preload all 8 A fragments (16 back-to-back ds_load_b128): lets each
    //      WMMA wait only for its own pair (staggered dscnt) and overlap the rest
    v16h afrag[8];
#pragma unroll
    for (int c = 0; c < 8; ++c) {
      const int ka = T0 + 32 * c;
      v8h alo = *(const v8h*)&H[arow * HSTRIDE + ka + akoff];
      v8h ahi = *(const v8h*)&H[arow * HSTRIDE + ka + 16 + akoff];
#pragma unroll
      for (int j = 0; j < 8; ++j) { afrag[c][j] = alo[j]; afrag[c][8 + j] = ahi[j]; }
    }

    // ---- oldpart(16x16) = Hist(16x256) @ Bshift(256x16) via 8x WMMA f16
    v8f acc = {};
#pragma unroll
    for (int c = 0; c < 8; ++c) {
      acc = __builtin_amdgcn_wmma_f32_16x16x32_f16(
          /*neg_a=*/false, afrag[c], /*neg_b=*/false, bfrag[c],
          /*c_mod=*/(short)0, acc, /*reuse_a=*/false, /*reuse_b=*/false);
    }

    // ---- spill C tile: VGPR r -> (M = r or r+8, N = lane%16)
#pragma unroll
    for (int r = 0; r < 8; ++r) {
      int m = (lane < 16) ? r : (r + 8);
      oldp[m * OSTRIDE + (lane & 15)] = acc[r];
    }
    __syncthreads();

    // ---- serial tail: lane s = batch b0+s; 16 steps, <=15 new-spike FMAs each
    if (lane < 16) {
      float sp[BLK];
#pragma unroll
      for (int tau = 0; tau < BLK; ++tau) {
        int t  = T0 + tau;
        int tc = (t < LENG) ? t : (LENG - 1);   // last block has 2 valid steps
        float fbv = oldp[lane * OSTRIDE + tau];
#pragma unroll
        for (int j = 0; j < tau; ++j) fbv = fmaf(wt[15 - tau + j], sp[j], fbv);
        float gensig = fmaf(fs_b, ft[tc], bias0) + fbv;
        float spike  = 1.0f / (1.0f + __expf(-gensig));
        sp[tau] = spike;
        H[lane * HSTRIDE + NBF + t] = (_Float16)spike;   // feed future blocks
        if (t < TSIM) {
          size_t o = (size_t)(b0 + lane) * TSIM + t;
          gensig_ws[o] = gensig;
          spike_ws[o]  = spike;
        }
      }
    }
    __syncthreads();
  }
}

// ---------------------------------------------------------------------------
// Kernel C: broadcast deduped traces to all 256 repeats (bandwidth-bound).
// d_out = [out1: (128,256,1500)] ++ [out2: (128,256,1250)], float2 granularity.
// ---------------------------------------------------------------------------
__global__ __launch_bounds__(256) void expand_kernel(
    const float* __restrict__ init_sec,
    const float* __restrict__ spike_ws,
    const float* __restrict__ gensig_ws,
    f2* __restrict__ out)
{
  const unsigned N1_2 = (unsigned)BATCH * NREP * (NBINS / 2);          // 24,576,000
  const unsigned NT_2 = N1_2 + (unsigned)BATCH * NREP * (TSIM / 2);    // 45,056,000
  unsigned idx = blockIdx.x * 256u + threadIdx.x;
  if (idx >= NT_2) return;
  f2 v;
  if (idx < N1_2) {
    unsigned row = idx / (NBINS / 2);
    unsigned c2  = idx - row * (NBINS / 2);
    unsigned b   = row >> 8;                     // NREP == 256
    if (c2 < NBF / 2)
      v = *(const f2*)&init_sec[(size_t)b * NBF + 2 * c2];
    else
      v = *(const f2*)&spike_ws[(size_t)b * TSIM + 2 * c2 - NBF];
  } else {
    unsigned j   = idx - N1_2;
    unsigned row = j / (TSIM / 2);
    unsigned c2  = j - row * (TSIM / 2);
    unsigned b   = row >> 8;
    v = *(const f2*)&gensig_ws[(size_t)b * TSIM + 2 * c2];
  }
  __builtin_nontemporal_store(v, &out[idx]);     // streaming: don't pollute L2
}

// ---------------------------------------------------------------------------
extern "C" void kernel_launch(void* const* d_in, const int* in_sizes, int n_in,
                              void* d_out, int out_size, void* d_ws, size_t ws_size,
                              hipStream_t stream) {
  const float* X       = (const float*)d_in[0];  // (128, 2048)
  const float* stim_t  = (const float*)d_in[1];  // (1500,)
  const float* init_s  = (const float*)d_in[2];  // (128, 250)
  // d_in[3] = n_repeats (int scalar) -> fixed at 256 per reference setup
  const float* w_spat  = (const float*)d_in[4];  // (2048,)
  const float* w_time  = (const float*)d_in[5];  // (250,)
  const float* w_fb    = (const float*)d_in[6];  // (250,)
  const float* bias    = (const float*)d_in[7];  // (1,)

  char*  ws        = (char*)d_ws;
  float* ft        = (float*)(ws);                                    // 1251 f32
  float* fs        = (float*)(ws + 5120);                             // 128 f32
  float* spike_ws  = (float*)(ws + 6144);                             // 128*1250 f32
  float* gensig_ws = (float*)(ws + 6144 + (size_t)BATCH * TSIM * 4);  // 128*1250 f32

  precompute_kernel<<<6, 256, 0, stream>>>(stim_t, w_time, X, w_spat, ft, fs);
  sim_kernel<<<BATCH / 16, 32, 0, stream>>>(init_s, w_fb, ft, fs, bias,
                                            spike_ws, gensig_ws);
  const unsigned total2 = (unsigned)BATCH * NREP * (NBINS / 2)
                        + (unsigned)BATCH * NREP * (TSIM / 2);
  expand_kernel<<<(total2 + 255) / 256, 256, 0, stream>>>(
      init_s, spike_ws, gensig_ws, (f2*)d_out);
  (void)in_sizes; (void)n_in; (void)out_size; (void)ws_size;
}
